// GPT2Block_5609227288854
// MI455X (gfx1250) — matrix operand order repaired and checked by reading, not compiled, not run on hardware
//
#include <hip/hip_runtime.h>
#include <hip/hip_bf16.h>
#include <cstdint>
#include <cstddef>

// ---------------- problem constants ----------------
#define DIMX   1024
#define NHEAD  16
#define DHEAD  64
#define SEQ    4096
#define BATCH  4
#define NTOK   (BATCH * SEQ)          // 16384 tokens
#define RMSEPS 1.1920929e-07f
#define EPSLIN 1e-6f
#define THETA_ 10000.0f

typedef unsigned short U16;
typedef __attribute__((ext_vector_type(16))) __bf16          v16bf;
typedef __attribute__((ext_vector_type(8)))  float           v8f;
typedef __attribute__((ext_vector_type(4)))  unsigned int    u32x4;
typedef __attribute__((ext_vector_type(4)))  unsigned short  us4;
typedef __attribute__((ext_vector_type(4)))  unsigned int    su4;  // D# group0
typedef __attribute__((ext_vector_type(8)))  int             si8;  // D# group1
typedef __attribute__((ext_vector_type(4)))  int             si4;

// ---------------- bf16 helpers ----------------
__device__ __forceinline__ U16 f2bf(float f) {
    unsigned int u = __float_as_uint(f);
    u += 0x7FFFu + ((u >> 16) & 1u);          // round-to-nearest-even
    return (U16)(u >> 16);
}
__device__ __forceinline__ float bf2f(U16 h) {
    return __uint_as_float(((unsigned int)h) << 16);
}

// Assemble a 16-element bf16 WMMA fragment from two 16-byte pieces.
union FragU { u32x4 q[2]; v16bf v; };
__device__ __forceinline__ v16bf ld_frag(const U16* p0, const U16* p1) {
    FragU f;
    f.q[0] = *(const u32x4*)p0;
    f.q[1] = *(const u32x4*)p1;
    return f.v;
}

__device__ __forceinline__ v8f wmma_bf16(v16bf a, v16bf b, v8f c) {
    // D = A(16x32) * B(32x16) + C, f32 accumulate
    return __builtin_amdgcn_wmma_f32_16x16x32_bf16(
        /*neg_a=*/false, a, /*neg_b=*/false, b,
        /*c_mod=*/(short)0, c, /*reuse_a=*/false, /*reuse_b=*/false);
}

// -------- CDNA5 async global->LDS copy (32 contiguous bytes per lane) -----
// LDS byte address = low 32 bits of the generic pointer.
__device__ __forceinline__ unsigned lds_addr_of(const void* p) {
    return (unsigned)(unsigned long long)p;
}
__device__ __forceinline__ void async_cp32B(unsigned lds, const U16* g) {
    asm volatile("global_load_async_to_lds_b128 %0, %1, off\n\t"
                 "global_load_async_to_lds_b128 %0, %1, off offset:16"
                 :: "v"(lds), "v"(g) : "memory");
}
template <int N>
__device__ __forceinline__ void wait_asynccnt() {
    asm volatile("s_wait_asynccnt %0" :: "n"(N) : "memory");
}
template <int N>
__device__ __forceinline__ void wait_tensorcnt() {
#if __has_builtin(__builtin_amdgcn_s_wait_tensorcnt)
    __builtin_amdgcn_s_wait_tensorcnt((short)N);
#else
    asm volatile("s_wait_tensorcnt %0" :: "n"(N) : "memory");
#endif
}

// -------- Tensor Data Mover: load a [128 rows x 64B] tile into LDS --------
// D# group1 (constant per kernel): data_size=2B; pad_enable, pad every
// 16 DWORDs (64B row) by 4 DWORDs (16B) -> LDS row pitch 80B == LDT elems;
// tile_dim0=32 elems, tile_dim1=128 rows; tensor dims huge (no OOB clamp);
// tensor_dim0_stride = K elems.
__device__ __forceinline__ si8 tdm_group1(int K) {
    si8 g;
    g[0] = (int)((1u << 16) | (1u << 20) | (3u << 22) | (3u << 25));
    g[1] = (int)0xFFFF0000u;          // tensor_dim0 lo16 -> [31:16]
    g[2] = (int)0xFFFF000Fu;          // tensor_dim0 hi / tensor_dim1 lo
    g[3] = (int)((32u << 16) | 0xFu); // tensor_dim1 hi | tile_dim0=32
    g[4] = 128;                       // tile_dim1=128, tile_dim2=0
    g[5] = K;                         // tensor_dim0_stride (data_size units)
    g[6] = 0;
    g[7] = 0;
    return g;
}
__device__ __forceinline__ su4 tdm_group0(unsigned lds, const U16* gptr) {
    unsigned long long ga = (unsigned long long)gptr;
    su4 g;
    g[0] = 1u;                                   // count=1, user descriptor
    g[1] = lds;                                  // lds_addr
    g[2] = (unsigned)ga;                         // global_addr[31:0]
    g[3] = (unsigned)((ga >> 32) & 0x01FFFFFFu) | (2u << 30); // addr hi | type=2
    return g;
}
__device__ __forceinline__ void tdm_load(su4 g0, si8 g1) {
#if __has_builtin(__builtin_amdgcn_tensor_load_to_lds)
#if defined(__clang_major__) && (__clang_major__ >= 23)
    si4 z4 = {0, 0, 0, 0};
    si8 z8 = {0, 0, 0, 0, 0, 0, 0, 0};
    __builtin_amdgcn_tensor_load_to_lds(g0, g1, z4, z4, z8, 0);
#else
    si4 z4 = {0, 0, 0, 0};
    __builtin_amdgcn_tensor_load_to_lds(g0, g1, z4, z4, 0);
#endif
#else
    asm volatile("tensor_load_to_lds %0, %1" :: "s"(g0), "s"(g1) : "memory");
#endif
}

// ---------------- kernel 0: f32 -> bf16 convert ----------------
__global__ void cvt_f32_bf16(const float* __restrict__ in, U16* __restrict__ out, int n) {
    int i = blockIdx.x * blockDim.x + threadIdx.x;
    if (i < n) out[i] = f2bf(in[i]);
}

// ---------------- kernel 1/5: 128x128-tile bf16 GEMM, C = A*B^T + bias ----
// A: [M][K] bf16 row-major, B: [N][K] bf16 row-major, C: [M][N] bf16.
// 256 threads = 8 waves; wave w owns a 16-row strip, 8 n-tiles (8 v8f accums).
// Double-buffered LDS: A tile staged by the Tensor Data Mover (TENSORcnt),
// B tile staged by per-lane async global->LDS (ASYNCcnt).
#define LDT 40   // LDS row stride (elements) == 64B row + 16B TDM pad
__global__ __launch_bounds__(256) void gemm_bf16_bias(
    const U16* __restrict__ A, const U16* __restrict__ B,
    const float* __restrict__ bias, U16* __restrict__ C,
    int M, int N, int K) {
    __shared__ U16 lA[2][128 * LDT];
    __shared__ U16 lB[2][128 * LDT];
    const int t    = threadIdx.x;
    const int lane = t & 31, wave = t >> 5;
    const int hl   = lane >> 4, ln = lane & 15;
    const int m0   = blockIdx.y * 128, n0 = blockIdx.x * 128;
    const int ldr  = t & 127;           // B loader row
    const int ldp  = t >> 7;            // B loader part (0/1): 16 elems each

    v8f acc[8];
    {
        v8f z = {0.f, 0.f, 0.f, 0.f, 0.f, 0.f, 0.f, 0.f};
        for (int j = 0; j < 8; ++j) acc[j] = z;
    }

    const U16* gAt = A + (size_t)m0 * K;                       // TDM tile base
    const U16* gB  = B + (size_t)(n0 + ldr) * K + ldp * 16;    // async per-lane
    const unsigned ldsA[2] = { lds_addr_of(&lA[0][0]), lds_addr_of(&lA[1][0]) };
    unsigned sB[2];
    sB[0] = lds_addr_of(&lB[0][ldr * LDT + ldp * 16]);
    sB[1] = lds_addr_of(&lB[1][ldr * LDT + ldp * 16]);
    const si8 g1 = tdm_group1(K);

    const int nk = K >> 5;              // K chunks of 32
    // prologue: stage chunk 0 into buffer 0
    if (t < 32) tdm_load(tdm_group0(ldsA[0], gAt), g1);
    async_cp32B(sB[0], gB);

    for (int s = 0; s < nk; ++s) {
        const int buf = s & 1;
        if (s + 1 < nk) {               // stage next chunk into other buffer
            if (t < 32) tdm_load(tdm_group0(ldsA[buf ^ 1], gAt + (s + 1) * 32), g1);
            async_cp32B(sB[buf ^ 1], gB + (s + 1) * 32);
            wait_tensorcnt<1>();        // chunk-s A tile resident
            wait_asynccnt<2>();         // chunk-s B tile resident
        } else {
            wait_tensorcnt<0>();
            wait_asynccnt<0>();
        }
        __syncthreads();                // all waves see chunk-s LDS data

        // A fragment: k pieces at 8*hl and 16+8*hl within the 32-chunk
        const U16* ar = &lA[buf][(wave * 16 + ln) * LDT];
        v16bf af = ld_frag(ar + 8 * hl, ar + 16 + 8 * hl);
        // software-pipelined B fragments: load j+1 before WMMA j
        const U16* b0 = &lB[buf][ln * LDT + 8 * hl];
        v16bf bcur = ld_frag(b0, b0 + 16);
        #pragma unroll
        for (int j = 0; j < 8; ++j) {
            v16bf bnext;
            if (j < 7) {
                const U16* br = &lB[buf][((j + 1) * 16 + ln) * LDT + 8 * hl];
                bnext = ld_frag(br, br + 16);
            }
            acc[j] = wmma_bf16(af, bcur, acc[j]);
            bcur = bnext;
        }
        __syncthreads();                // done reading buf before it's refilled
    }

    // C/D layout: lane ln = n; vgpr v -> m = v + 8*hl
    for (int j = 0; j < 8; ++j) {
        int n = n0 + j * 16 + ln;
        float bs = bias[n];
        for (int v = 0; v < 8; ++v) {
            int m = m0 + wave * 16 + v + 8 * hl;
            C[(size_t)m * N + n] = f2bf(acc[j][v] + bs);
        }
    }
}

// ---------------- kernel 2: rmsnorm(q,k) + RoPE + ReLU + relayout --------
// qkv: [NTOK][3*DIM] bf16. Outputs:
//   qf : [64(bh)][SEQ][64]  bf16  (row-major in d)
//   kfT: [64(bh)][64][SEQ]  bf16  (transposed: row-major in l)
//   vT : [64(bh)][80][SEQ]  bf16  (row p=64 is ones; 65..79 zeros)
__global__ __launch_bounds__(256) void norm_rope_kernel(
    const U16* __restrict__ qkv,
    const float* __restrict__ gq, const float* __restrict__ gk,
    U16* __restrict__ qf, U16* __restrict__ kfT, U16* __restrict__ vT) {
    const int token = blockIdx.x;
    const int b = token >> 12, l = token & 4095;
    const int t = threadIdx.x;
    const int i0 = t * 4;                      // 4 contiguous dims = 2 rope pairs

    const U16* row = qkv + (size_t)token * (3 * DIMX);
    us4 uq = *(const us4*)(row + i0);
    us4 uk = *(const us4*)(row + DIMX + i0);
    us4 uv = *(const us4*)(row + 2 * DIMX + i0);
    float q[4], k[4], v[4];
    float sq = 0.f, sk = 0.f;
    for (int j = 0; j < 4; ++j) {
        q[j] = bf2f(uq[j]); k[j] = bf2f(uk[j]); v[j] = bf2f(uv[j]);
        sq += q[j] * q[j];  sk += k[j] * k[j];
    }

    __shared__ float redq[256], redk[256];
    redq[t] = sq; redk[t] = sk;
    __syncthreads();
    for (int s = 128; s > 0; s >>= 1) {
        if (t < s) { redq[t] += redq[t + s]; redk[t] += redk[t + s]; }
        __syncthreads();
    }
    const float rq = rsqrtf(redq[0] * (1.0f / DIMX) + RMSEPS);
    const float rk = rsqrtf(redk[0] * (1.0f / DIMX) + RMSEPS);

    const int h  = i0 >> 6;                   // constant over the 4 elems
    const int bh = b * NHEAD + h;
    for (int pp = 0; pp < 2; ++pp) {
        int i = i0 + 2 * pp;                  // even lane of the pair
        int d = i & 63;
        int p = d >> 1;                       // freq index 0..31
        float inv = __powf(THETA_, -(float)p * (1.0f / 32.0f));
        float ang = (float)l * inv;
        float sn, cs;
        __sincosf(ang, &sn, &cs);
        float qe = q[2 * pp] * rq * gq[i], qo = q[2 * pp + 1] * rq * gq[i + 1];
        float ke = k[2 * pp] * rk * gk[i], ko = k[2 * pp + 1] * rk * gk[i + 1];
        float qr = fmaxf(qe * cs - qo * sn, 0.f);
        float qi = fmaxf(qe * sn + qo * cs, 0.f);
        float kr = fmaxf(ke * cs - ko * sn, 0.f);
        float ki = fmaxf(ke * sn + ko * cs, 0.f);

        size_t qb = ((size_t)bh * SEQ + l) * 64 + d;
        qf[qb]     = f2bf(qr);
        qf[qb + 1] = f2bf(qi);
        size_t kb = ((size_t)bh * 64 + d) * SEQ + l;
        kfT[kb]        = f2bf(kr);
        kfT[kb + SEQ]  = f2bf(ki);
        size_t vb = ((size_t)bh * 80 + d) * SEQ + l;
        vT[vb]       = f2bf(v[2 * pp]);
        vT[vb + SEQ] = f2bf(v[2 * pp + 1]);
    }
    // vpad ones row (p=64) and zero the pad rows 65..79
    if (t < NHEAD)
        vT[((size_t)(b * NHEAD + t) * 80 + 64) * SEQ + l] = (U16)0x3F80; // bf16(1.0)
    if (t < 240) {
        int hh = t / 15, pz = 65 + t % 15;
        vT[((size_t)(b * NHEAD + hh) * 80 + pz) * SEQ + l] = 0;
    }
}

// ---------------- kernel 3: VK[p][d] = sum_l vpad[l][p] * kf[l][d] -------
// grid (5, 64): x = p-tile, y = bh. 128 threads = 4 waves; wave = d-tile.
__global__ __launch_bounds__(128) void vk_kernel(
    const U16* __restrict__ vT, const U16* __restrict__ kfT,
    U16* __restrict__ VK) {
    const int bh = blockIdx.y, mt = blockIdx.x;
    const int t = threadIdx.x, lane = t & 31, wave = t >> 5;
    const int hl = lane >> 4, ln = lane & 15;

    const U16* vb = vT  + (size_t)bh * 80 * SEQ + (size_t)(mt * 16 + ln) * SEQ;
    const U16* kb = kfT + (size_t)bh * 64 * SEQ + (size_t)(wave * 16 + ln) * SEQ;

    v8f acc = {0.f, 0.f, 0.f, 0.f, 0.f, 0.f, 0.f, 0.f};
    for (int kk = 0; kk < SEQ; kk += 32) {
        if (kk + 32 < SEQ) {                  // gfx1250 global_prefetch_b8
            __builtin_prefetch(vb + kk + 32, 0, 1);
            __builtin_prefetch(kb + kk + 32, 0, 1);
        }
        v16bf a  = ld_frag(vb + kk + 8 * hl, vb + kk + 16 + 8 * hl);
        v16bf bf = ld_frag(kb + kk + 8 * hl, kb + kk + 16 + 8 * hl);
        acc = wmma_bf16(a, bf, acc);
    }
    U16* out = VK + (size_t)bh * 80 * 64;
    for (int v = 0; v < 8; ++v) {
        int m = mt * 16 + v + 8 * hl;      // p
        int n = wave * 16 + ln;            // d
        out[m * 64 + n] = f2bf(m < 65 ? acc[v] : 0.0f);
    }
}

// ---------------- kernel 4: res = qf @ VK^T, divide, relayout ------------
// grid (SEQ/128, 64): y = bh. 256 threads = 8 waves; wave = 16-row l-tile.
__global__ __launch_bounds__(256) void res_kernel(
    const U16* __restrict__ qf, const U16* __restrict__ VK,
    U16* __restrict__ attn) {
    const int bh = blockIdx.y, b = bh >> 4, h = bh & 15;
    const int t = threadIdx.x, lane = t & 31, wave = t >> 5;
    const int hl = lane >> 4, ln = lane & 15;
    const int ltile = blockIdx.x * 128 + wave * 16;

    const U16* qb  = qf + ((size_t)bh * SEQ + ltile + ln) * 64;
    const U16* vkb = VK + (size_t)bh * 80 * 64;

    v8f acc[5];
    {
        v8f z = {0.f, 0.f, 0.f, 0.f, 0.f, 0.f, 0.f, 0.f};
        for (int j = 0; j < 5; ++j) acc[j] = z;
    }
    for (int ks = 0; ks < 64; ks += 32) {
        v16bf a = ld_frag(qb + ks + 8 * hl, qb + ks + 16 + 8 * hl);
        for (int j = 0; j < 5; ++j) {
            const U16* br = vkb + (j * 16 + ln) * 64;
            v16bf bf = ld_frag(br + ks + 8 * hl, br + ks + 16 + 8 * hl);
            acc[j] = wmma_bf16(a, bf, acc[j]);
        }
    }
    // denominator: accum 4, column n==0 of this lane's row half (wave32 shfl)
    for (int v = 0; v < 8; ++v) {
        float denom = __shfl(acc[4][v], lane & 16, 32) + EPSLIN;
        int grow = b * SEQ + ltile + v + 8 * hl;
        for (int j = 0; j < 4; ++j) {
            int col = h * 64 + j * 16 + ln;
            attn[(size_t)grow * DIMX + col] = f2bf(acc[j][v] / denom);
        }
    }
}

// ---------------- kernel 6: final rmsnorm, f32 out ------------------------
__global__ __launch_bounds__(256) void final_norm_kernel(
    const U16* __restrict__ tmp, const float* __restrict__ gout,
    float* __restrict__ out) {
    const int token = blockIdx.x, t = threadIdx.x;
    const U16* row = tmp + (size_t)token * DIMX;
    us4 u = *(const us4*)(row + t * 4);
    float x[4];
    float s = 0.f;
    for (int j = 0; j < 4; ++j) { x[j] = bf2f(u[j]); s += x[j] * x[j]; }
    __shared__ float red[256];
    red[t] = s;
    __syncthreads();
    for (int sft = 128; sft > 0; sft >>= 1) {
        if (t < sft) red[t] += red[t + sft];
        __syncthreads();
    }
    float r = rsqrtf(red[0] * (1.0f / DIMX) + RMSEPS);
    for (int j = 0; j < 4; ++j)
        out[(size_t)token * DIMX + t * 4 + j] = x[j] * r * gout[t * 4 + j];
}

// ---------------- host launcher ------------------------------------------
extern "C" void kernel_launch(void* const* d_in, const int* in_sizes, int n_in,
                              void* d_out, int out_size, void* d_ws, size_t ws_size,
                              hipStream_t stream) {
    (void)in_sizes; (void)n_in; (void)out_size; (void)ws_size;
    const float* x    = (const float*)d_in[0];
    const float* Wqkv = (const float*)d_in[1];
    const float* bqkv = (const float*)d_in[2];
    const float* gq   = (const float*)d_in[3];
    const float* gk   = (const float*)d_in[4];
    const float* Wout = (const float*)d_in[5];
    const float* bout = (const float*)d_in[6];
    const float* gout = (const float*)d_in[7];

    char* ws = (char*)d_ws;
    size_t off = 0;
    auto take = [&](size_t bytes) -> char* {
        char* p = ws + off;
        off = (off + bytes + 255) & ~(size_t)255;
        return p;
    };
    U16* xb    = (U16*)take((size_t)NTOK * DIMX * 2);
    U16* wqkvb = (U16*)take((size_t)3 * DIMX * DIMX * 2);
    U16* woutb = (U16*)take((size_t)DIMX * DIMX * 2);
    U16* qkvb  = (U16*)take((size_t)NTOK * 3 * DIMX * 2);
    U16* qfb   = (U16*)take((size_t)64 * SEQ * 64 * 2);
    U16* kftb  = (U16*)take((size_t)64 * 64 * SEQ * 2);
    U16* vtb   = (U16*)take((size_t)64 * 80 * SEQ * 2);
    U16* vkb   = (U16*)take((size_t)64 * 80 * 64 * 2);
    U16* attnb = (U16*)take((size_t)NTOK * DIMX * 2);
    U16* tmpb  = (U16*)take((size_t)NTOK * DIMX * 2);

    int n1 = NTOK * DIMX;
    cvt_f32_bf16<<<(n1 + 255) / 256, 256, 0, stream>>>(x, xb, n1);
    int n2 = 3 * DIMX * DIMX;
    cvt_f32_bf16<<<(n2 + 255) / 256, 256, 0, stream>>>(Wqkv, wqkvb, n2);
    int n3 = DIMX * DIMX;
    cvt_f32_bf16<<<(n3 + 255) / 256, 256, 0, stream>>>(Wout, woutb, n3);

    // qkv = x @ Wqkv^T + bqkv
    gemm_bf16_bias<<<dim3(3 * DIMX / 128, NTOK / 128), 256, 0, stream>>>(
        xb, wqkvb, bqkv, qkvb, NTOK, 3 * DIMX, DIMX);

    norm_rope_kernel<<<NTOK, 256, 0, stream>>>(qkvb, gq, gk, qfb, kftb, vtb);

    vk_kernel<<<dim3(5, 64), 128, 0, stream>>>(vtb, kftb, vkb);

    res_kernel<<<dim3(SEQ / 128, 64), 256, 0, stream>>>(qfb, vkb, attnb);

    // out1 = attn @ Wout^T + bout
    gemm_bf16_bias<<<dim3(DIMX / 128, NTOK / 128), 256, 0, stream>>>(
        attnb, woutb, bout, tmpb, NTOK, DIMX, DIMX);

    final_norm_kernel<<<NTOK, 256, 0, stream>>>(tmpb, gout, (float*)d_out);
}
